// MultiHeadSelfAttention_22325240004791
// MI455X (gfx1250) — matrix operand order repaired and checked by reading, not compiled, not run on hardware
//
#include <hip/hip_runtime.h>
#include <hip/hip_bf16.h>
#include <math.h>

typedef __bf16 bf16;
typedef __attribute__((ext_vector_type(16))) __bf16 v16bf;
typedef __attribute__((ext_vector_type(8)))  __bf16 v8bf;
typedef __attribute__((ext_vector_type(8)))  float  v8f;
typedef uint32_t u32x4 __attribute__((ext_vector_type(4)));
typedef uint32_t u32x8 __attribute__((ext_vector_type(8)));

#define BB 2
#define TT 2048
#define DD 1024
#define HH 16
#define HDIM 64
#define MM (BB*TT)   /* 4096 rows of x */

// ---------------------------------------------------------------------------
// f32 -> bf16 conversion (grid-stride)
// ---------------------------------------------------------------------------
__global__ void cvt_f32_to_bf16(const float* __restrict__ in,
                                bf16* __restrict__ out, int n) {
    int i = blockIdx.x * blockDim.x + threadIdx.x;
    int stride = gridDim.x * blockDim.x;
    for (; i < n; i += stride) out[i] = (bf16)in[i];
}

// ---------------------------------------------------------------------------
// GEMM: C[M,N] = A[M,K] * W[N,K]^T   (M=4096, N=K=1024)
// One wave computes a 32x64 tile: 8 accumulators, 2 A frags, 4 B frags
// (each B frag reused across both M halves => ~21 FLOP/B from cache).
// Row-major W rows ARE WMMA B fragments for x @ W^T (contiguous 32B loads).
// mode 0: bf16 row-major [M,N]; mode 1: bf16 V^T [B,H,HD,T]; mode 2: f32 [M,N]
// ---------------------------------------------------------------------------
__global__ void gemm_xwT(const bf16* __restrict__ A, const bf16* __restrict__ W,
                         bf16* __restrict__ outb, float* __restrict__ outf,
                         int mode) {
    const int lane = threadIdx.x & 31;
    const int wave = threadIdx.x >> 5;
    const int half = lane >> 4;
    const int lidx = lane & 15;
    const int n0 = blockIdx.x * 64;
    const int m0 = blockIdx.y * 128 + wave * 32;

    v8f acc0[4] = {};
    v8f acc1[4] = {};
    const bf16* arow0 = A + (size_t)(m0 + lidx) * DD;
    const bf16* arow1 = A + (size_t)(m0 + 16 + lidx) * DD;

    for (int k = 0; k < DD; k += 32) {
        if (k + 32 < DD) {
            __builtin_prefetch(arow0 + k + 32, 0, 1);
            __builtin_prefetch(arow1 + k + 32, 0, 1);
        }
        v16bf a0, a1;
        {
            v8bf lo0 = *(const v8bf*)(arow0 + k + half * 16);
            v8bf hi0 = *(const v8bf*)(arow0 + k + 8 + half * 16);
            v8bf lo1 = *(const v8bf*)(arow1 + k + half * 16);
            v8bf hi1 = *(const v8bf*)(arow1 + k + 8 + half * 16);
#pragma unroll
            for (int i = 0; i < 8; ++i) {
                a0[i] = lo0[i]; a0[8 + i] = hi0[i];
                a1[i] = lo1[i]; a1[8 + i] = hi1[i];
            }
        }
#pragma unroll
        for (int n = 0; n < 4; ++n) {
            const bf16* wrow = W + (size_t)(n0 + n * 16 + lidx) * DD + k + half * 16;
            v16bf bw = *(const v16bf*)wrow;
            acc0[n] = __builtin_amdgcn_wmma_f32_16x16x32_bf16(
                false, a0, false, bw, (short)0, acc0[n], false, false);
            acc1[n] = __builtin_amdgcn_wmma_f32_16x16x32_bf16(
                false, a1, false, bw, (short)0, acc1[n], false, false);
        }
    }

#pragma unroll
    for (int hfm = 0; hfm < 2; ++hfm) {
#pragma unroll
        for (int n = 0; n < 4; ++n) {
#pragma unroll
            for (int r = 0; r < 8; ++r) {
                int m  = m0 + hfm * 16 + r + 8 * half;
                int gn = n0 + n * 16 + lidx;
                float v = hfm ? acc1[n][r] : acc0[n][r];
                if (mode == 0) {
                    outb[(size_t)m * DD + gn] = (bf16)v;
                } else if (mode == 1) {
                    int b = m / TT, t = m % TT;
                    int h = gn / HDIM, hd = gn % HDIM;
                    outb[((size_t)((b * HH + h) * HDIM + hd)) * TT + t] = (bf16)v;
                } else {
                    outf[(size_t)m * DD + gn] = v;
                }
            }
        }
    }
}

// ---------------------------------------------------------------------------
// Flash attention (causal): one wave per (b, h, 16-row q tile).
// Per 32-key block:
//   K tile (32x64, stride DD)  -> LDS via TDM  tensor_load_to_lds (TENSORcnt)
//   V^T tile (64x32, stride T) -> LDS via GLOBAL_LOAD_ASYNC_TO_LDS (ASYNCcnt)
// double-buffered so the DMA of block kb+1 overlaps compute of block kb.
// Then 4 WMMAs for S = Q K^T, online softmax (f32), P relayout through LDS,
// 4 WMMAs for O += P V.
// ---------------------------------------------------------------------------
__global__ void attn_fwd(const bf16* __restrict__ Q, const bf16* __restrict__ K,
                         const bf16* __restrict__ VT, bf16* __restrict__ CTX) {
    const int lane = threadIdx.x & 31;
    const int half = lane >> 4;
    const int lidx = lane & 15;
    const int qt = blockIdx.x;  // q tile (16 rows)
    const int h  = blockIdx.y;
    const int b  = blockIdx.z;

    __shared__ __align__(128) bf16 klds[2 * 32 * 64];  // [buf][key][hd] 2x4KB
    __shared__ __align__(128) bf16 vlds[2 * 64 * 32];  // [buf][hd][key] 2x4KB
    __shared__ __align__(128) bf16 plds[16 * 32];      // P relayout      1KB

    const uint32_t klds_base = (uint32_t)(uintptr_t)&klds[0];
    const uint32_t vlds_base = (uint32_t)(uintptr_t)&vlds[0];

    // Issue DMA for key-block kbv into LDS buffer buf.
    auto issue_tile = [&](int kbv, int buf) {
        // --- K tile via Tensor Data Mover: 2D tile 64(hd) x 32(keys)
        uint64_t kg = (uint64_t)(uintptr_t)(K + ((size_t)(b * TT) + kbv * 32) * DD
                                              + h * HDIM);
        u32x4 g0;
        u32x8 g1;
        g0[0] = 1u;                                     // count=1, user mode
        g0[1] = klds_base + (uint32_t)buf * 4096u;      // lds_addr (bytes)
        g0[2] = (uint32_t)kg;                           // global_addr[31:0]
        g0[3] = (uint32_t)(kg >> 32) | 0x80000000u;     // addr[56:32] | type=2
        g1[0] = 0x00010000u;       // workgroup_mask=0, data_size=1 (2 bytes)
        g1[1] = 64u << 16;         // tensor_dim0 = 64 (bits 63:48)
        g1[2] = 2048u << 16;       // tensor_dim1[15:0] = 2048 (bits 95:80)
        g1[3] = 64u << 16;         // tile_dim0 = 64 (bits 127:112)
        g1[4] = 32u;               // tile_dim1 = 32, tile_dim2 = 0
        g1[5] = 1024u;             // tensor_dim0_stride = DD (elements)
        g1[6] = 0u;                // stride hi / dim1_stride lo
        g1[7] = 0u;
        asm volatile("tensor_load_to_lds %0, %1" :: "s"(g0), "s"(g1) : "memory");

        // --- V^T tile via async-to-LDS: 64 rows x 64 B, row stride T*2
        const char* vbase =
            (const char*)(VT + ((size_t)((b * HH + h) * HDIM)) * TT + kbv * 32);
        uint32_t vdst = vlds_base + (uint32_t)buf * 4096u;
#pragma unroll
        for (int i = 0; i < 8; ++i) {
            uint32_t f = (uint32_t)(lane * 16) + (uint32_t)(i * 512);
            uint32_t vrow = f >> 6, vcol = f & 63u;
            uint64_t vg = (uint64_t)(uintptr_t)(vbase + (size_t)vrow * (TT * 2) + vcol);
            uint32_t vl = vdst + f;
            asm volatile("global_load_async_to_lds_b128 %0, %1, off"
                         :: "v"(vl), "v"(vg) : "memory");
        }
    };

    // Q fragments: 16 x 64 = two 16x32 A fragments, loaded once
    const bf16* qrow = Q + ((size_t)(b * TT) + qt * 16 + lidx) * DD + h * HDIM;
    v16bf qf[2];
#pragma unroll
    for (int kk = 0; kk < 2; ++kk) {
        v8bf lo = *(const v8bf*)(qrow + kk * 32 + half * 16);
        v8bf hi = *(const v8bf*)(qrow + kk * 32 + 8 + half * 16);
#pragma unroll
        for (int i = 0; i < 8; ++i) { qf[kk][i] = lo[i]; qf[kk][8 + i] = hi[i]; }
    }

    v8f acc[4] = {};
    float m_i[8], l_i[8];
#pragma unroll
    for (int r = 0; r < 8; ++r) { m_i[r] = -INFINITY; l_i[r] = 0.f; }

    const float scale = 0.125f;             // 1/sqrt(64)
    const int kbmax = (qt * 16 + 15) / 32;  // causal truncation of key loop

    issue_tile(0, 0);  // prologue: stage first key block

    for (int kb = 0; kb <= kbmax; ++kb) {
        const int buf = kb & 1;
        // wait for this block's DMA, then immediately start the next block's
        asm volatile("s_wait_tensorcnt 0x0" ::: "memory");
        asm volatile("s_wait_asynccnt 0x0" ::: "memory");
        if (kb + 1 <= kbmax) issue_tile(kb + 1, buf ^ 1);

        const bf16* kt = &klds[buf * 2048];
        const bf16* vt = &vlds[buf * 2048];

        // ---- scores: S (16 x 32) as two 16x16 C fragments, K frags from LDS
        v8f s[2] = {};
#pragma unroll
        for (int c = 0; c < 2; ++c) {
#pragma unroll
            for (int kk = 0; kk < 2; ++kk) {
                v16bf bk = *(const v16bf*)&kt[(c * 16 + lidx) * 64 +
                                              kk * 32 + half * 16];
                s[c] = __builtin_amdgcn_wmma_f32_16x16x32_bf16(
                    false, qf[kk], false, bk, (short)0, s[c], false, false);
            }
        }

        // ---- scale + causal mask + online softmax (row stats per C-frag row)
        float p0s[8], p1s[8], alph[8];
#pragma unroll
        for (int r = 0; r < 8; ++r) {
            int row = qt * 16 + r + 8 * half;
            int c0 = kb * 32 + lidx;
            int c1 = c0 + 16;
            float v0 = s[0][r] * scale;
            float v1 = s[1][r] * scale;
            if (c0 > row) v0 = -INFINITY;
            if (c1 > row) v1 = -INFINITY;
            float v = fmaxf(v0, v1);
#pragma unroll
            for (int off = 1; off < 16; off <<= 1)
                v = fmaxf(v, __shfl_xor(v, off, 16));
            float mn = fmaxf(m_i[r], v);
            float alpha = __expf(m_i[r] - mn);
            float p0 = __expf(v0 - mn);
            float p1 = __expf(v1 - mn);
            float ps = p0 + p1;
#pragma unroll
            for (int off = 1; off < 16; off <<= 1)
                ps += __shfl_xor(ps, off, 16);
            l_i[r] = l_i[r] * alpha + ps;
            m_i[r] = mn;
            p0s[r] = p0; p1s[r] = p1; alph[r] = alpha;
        }
#pragma unroll
        for (int n = 0; n < 4; ++n)
#pragma unroll
            for (int r = 0; r < 8; ++r)
                acc[n][r] *= alph[r];

        // ---- P relayout: C-frag (rows across lanes) -> A-frag (rows per lane)
#pragma unroll
        for (int r = 0; r < 8; ++r) {
            plds[(r + 8 * half) * 32 + lidx]      = (bf16)p0s[r];
            plds[(r + 8 * half) * 32 + 16 + lidx] = (bf16)p1s[r];
        }
        __syncthreads();
        v16bf pa;
        {
            v8bf lo = *(const v8bf*)&plds[lidx * 32 + half * 16];
            v8bf hi = *(const v8bf*)&plds[lidx * 32 + 8 + half * 16];
#pragma unroll
            for (int i = 0; i < 8; ++i) { pa[i] = lo[i]; pa[8 + i] = hi[i]; }
        }
        __syncthreads();

        // ---- O += P (16x32) * V (32x64): 4 WMMAs, B frags from LDS V^T tile
#pragma unroll
        for (int n = 0; n < 4; ++n) {
            v16bf bv = *(const v16bf*)&vt[(n * 16 + lidx) * 32 + half * 16];
            acc[n] = __builtin_amdgcn_wmma_f32_16x16x32_bf16(
                false, pa, false, bv, (short)0, acc[n], false, false);
        }
    }

    // ---- normalize and store context (bf16, [B,T,D] head-major within D)
#pragma unroll
    for (int r = 0; r < 8; ++r) {
        float inv = 1.f / l_i[r];
        int t = qt * 16 + r + 8 * half;
#pragma unroll
        for (int n = 0; n < 4; ++n) {
            CTX[((size_t)(b * TT) + t) * DD + h * HDIM + n * 16 + lidx] =
                (bf16)(acc[n][r] * inv);
        }
    }
}

// ---------------------------------------------------------------------------
extern "C" void kernel_launch(void* const* d_in, const int* in_sizes, int n_in,
                              void* d_out, int out_size, void* d_ws, size_t ws_size,
                              hipStream_t stream) {
    const float* x  = (const float*)d_in[0];
    const float* wq = (const float*)d_in[1];
    const float* wk = (const float*)d_in[2];
    const float* wv = (const float*)d_in[3];
    const float* wo = (const float*)d_in[4];
    float* out = (float*)d_out;

    const size_t nX = (size_t)BB * TT * DD;  // 4,194,304
    const size_t nW = (size_t)DD * DD;       // 1,048,576

    bf16* ws   = (bf16*)d_ws;
    bf16* xb   = ws;
    bf16* wqb  = xb  + nX;
    bf16* wkb  = wqb + nW;
    bf16* wvb  = wkb + nW;
    bf16* wob  = wvb + nW;
    bf16* qb   = wob + nW;
    bf16* kb   = qb  + nX;
    bf16* vtb  = kb  + nX;   // V stored transposed [B,H,HD,T]
    bf16* ctxb = vtb + nX;
    (void)ws_size; (void)n_in; (void)in_sizes; (void)out_size;

    // 1) convert to bf16
    cvt_f32_to_bf16<<<4096, 256, 0, stream>>>(x,  xb,  (int)nX);
    cvt_f32_to_bf16<<<1024, 256, 0, stream>>>(wq, wqb, (int)nW);
    cvt_f32_to_bf16<<<1024, 256, 0, stream>>>(wk, wkb, (int)nW);
    cvt_f32_to_bf16<<<1024, 256, 0, stream>>>(wv, wvb, (int)nW);
    cvt_f32_to_bf16<<<1024, 256, 0, stream>>>(wo, wob, (int)nW);

    // 2) projections (M=4096, N=K=1024); block = 4 waves, each wave 32x64 tile
    dim3 ggrid(DD / 64, MM / 128, 1);  // 16 x 32
    gemm_xwT<<<ggrid, 128, 0, stream>>>(xb, wqb, qb,  nullptr, 0);
    gemm_xwT<<<ggrid, 128, 0, stream>>>(xb, wkb, kb,  nullptr, 0);
    gemm_xwT<<<ggrid, 128, 0, stream>>>(xb, wvb, vtb, nullptr, 1);

    // 3) causal flash attention: (T/16) x H x B waves
    attn_fwd<<<dim3(TT / 16, HH, BB), 32, 0, stream>>>(qb, kb, vtb, ctxb);

    // 4) output projection -> f32
    gemm_xwT<<<ggrid, 128, 0, stream>>>(ctxb, wob, nullptr, out, 2);
}
